// SATemporalGATLayer_89026082111543
// MI455X (gfx1250) — matrix operand order repaired and checked by reading, not compiled, not run on hardware
//
#include <hip/hip_runtime.h>
#include <cstdint>

#define N_NODES 50000
#define E_EDGES (N_NODES + N_NODES * 16)
#define DIM 128
#define HEADS 8
#define CH 16

typedef __attribute__((ext_vector_type(16))) __bf16 v16bf;
typedef __attribute__((ext_vector_type(8)))  float  v8f;

union FragBF { v16bf v; unsigned u[8]; };

__device__ __forceinline__ unsigned short f2bf(float f) {
    unsigned u = __float_as_uint(f);
    u += 0x7FFFu + ((u >> 16) & 1u);   // round-to-nearest-even
    return (unsigned short)(u >> 16);
}

__device__ __forceinline__ void atomicMaxF(float* addr, float val) {
    // monotonic bit-pattern trick: int-max for >=0, uint-min for <0
    if (val >= 0.0f) atomicMax((int*)addr, __float_as_int(val));
    else             atomicMin((unsigned int*)addr, __float_as_uint(val));
}

// gfx1250 async global->LDS copy (GVS mode), tracked by ASYNCcnt
__device__ __forceinline__ void async_b128_to_lds(unsigned lds_off, unsigned goff,
                                                  const void* sbase) {
    asm volatile("global_load_async_to_lds_b128 %0, %1, %2"
                 :: "v"(lds_off), "v"(goff), "s"(sbase) : "memory");
}
__device__ __forceinline__ void wait_async0() {
    asm volatile("s_wait_asynccnt 0" ::: "memory");
}

// ---------------------------------------------------------------- init
__global__ void k_init(float* out, float* segmax, float* segsum) {
    int i = blockIdx.x * blockDim.x + threadIdx.x;
    if (i < N_NODES * DIM) out[i] = 0.0f;
    if (i < N_NODES * HEADS) { segsum[i] = 0.0f; segmax[i] = -__builtin_inff(); }
}

// ---------------------------------------------------------------- prep: wl/wr + bf16 W^T
__global__ void k_prep(const float* __restrict__ W, const float* __restrict__ Wres,
                       const float* __restrict__ attL, const float* __restrict__ attR,
                       float* wl, float* wr,
                       unsigned short* WTbf, unsigned short* WTresbf) {
    int d = threadIdx.x;                       // 0..127
    #pragma unroll
    for (int h = 0; h < HEADS; ++h) {
        float sl = 0.f, sr = 0.f;
        #pragma unroll
        for (int c = 0; c < CH; ++c) {
            float w = W[d * DIM + h * CH + c];
            sl += w * attL[h * CH + c];
            sr += w * attR[h * CH + c];
        }
        wl[d * HEADS + h] = sl;
        wr[d * HEADS + h] = sr;
    }
    for (int n = 0; n < DIM; ++n) {            // transposed (N-major) bf16 copies
        WTbf[n * DIM + d]    = f2bf(W[d * DIM + n]);
        WTresbf[n * DIM + d] = f2bf(Wres[d * DIM + n]);
    }
}

// ---------------------------------------------------------------- r[n,h] = filtered[n] . wr[:,h]
// 4 lanes per node: coalesced 64B/4-lane reads + shfl_xor butterfly reduce
__global__ void k_node_r(const float* __restrict__ ea, const float* __restrict__ wr,
                         float* __restrict__ r) {
    __shared__ float wrs[DIM * HEADS];
    for (int i = threadIdx.x; i < DIM * HEADS; i += blockDim.x) wrs[i] = wr[i];
    __syncthreads();
    int gid = blockIdx.x * blockDim.x + threadIdx.x;
    int n = gid >> 2, sub = gid & 3;
    if (n >= N_NODES) return;
    const float4* row = (const float4*)(ea + (size_t)n * DIM);
    float acc[HEADS] = {0.f, 0.f, 0.f, 0.f, 0.f, 0.f, 0.f, 0.f};
    #pragma unroll
    for (int q = sub; q < DIM / 4; q += 4) {
        float4 v = row[q];
        int d = q * 4;
        #pragma unroll
        for (int h = 0; h < HEADS; ++h)
            acc[h] += v.x * wrs[(d + 0) * HEADS + h] + v.y * wrs[(d + 1) * HEADS + h]
                    + v.z * wrs[(d + 2) * HEADS + h] + v.w * wrs[(d + 3) * HEADS + h];
    }
    #pragma unroll
    for (int h = 0; h < HEADS; ++h) {
        acc[h] += __shfl_xor(acc[h], 1);
        acc[h] += __shfl_xor(acc[h], 2);
    }
    if (sub == 0) {
        #pragma unroll
        for (int h = 0; h < HEADS; ++h) r[n * HEADS + h] = acc[h];
    }
}

// ---------------------------------------------------------------- alpha + leaky relu + segmax
__global__ void k_alpha(const float* __restrict__ ea, const float* __restrict__ wl,
                        const float* __restrict__ r, const int* __restrict__ tgt,
                        const int* __restrict__ src, float* __restrict__ alpha,
                        float* __restrict__ segmax) {
    __shared__ float wls[DIM * HEADS];
    for (int i = threadIdx.x; i < DIM * HEADS; i += blockDim.x) wls[i] = wl[i];
    __syncthreads();
    int gid = blockIdx.x * blockDim.x + threadIdx.x;
    int e = gid >> 2, sub = gid & 3;
    if (e >= E_EDGES) return;
    const float4* row = (const float4*)(ea + (size_t)e * DIM);
    float acc[HEADS] = {0.f, 0.f, 0.f, 0.f, 0.f, 0.f, 0.f, 0.f};
    #pragma unroll
    for (int q = sub; q < DIM / 4; q += 4) {
        float4 v = row[q];
        int d = q * 4;
        #pragma unroll
        for (int h = 0; h < HEADS; ++h)
            acc[h] += v.x * wls[(d + 0) * HEADS + h] + v.y * wls[(d + 1) * HEADS + h]
                    + v.z * wls[(d + 2) * HEADS + h] + v.w * wls[(d + 3) * HEADS + h];
    }
    #pragma unroll
    for (int h = 0; h < HEADS; ++h) {
        acc[h] += __shfl_xor(acc[h], 1);
        acc[h] += __shfl_xor(acc[h], 2);
    }
    if (sub == 0) {
        int t = tgt[e], s = src[e];
        #pragma unroll
        for (int h = 0; h < HEADS; ++h) {
            float a = acc[h] + r[t * HEADS + h];
            a = (a > 0.f) ? a : 0.2f * a;      // leaky_relu slope 0.2
            alpha[(size_t)e * HEADS + h] = a;
            atomicMaxF(&segmax[s * HEADS + h], a);
        }
    }
}

// ---------------------------------------------------------------- exp + segment sum
__global__ void k_ex(const int* __restrict__ src, float* __restrict__ alpha,
                     const float* __restrict__ segmax, float* __restrict__ segsum) {
    int i = blockIdx.x * blockDim.x + threadIdx.x;
    if (i >= E_EDGES * HEADS) return;
    int e = i >> 3, h = i & 7;
    int s = src[e];
    float x = __expf(alpha[i] - segmax[s * HEADS + h]);
    alpha[i] = x;
    atomicAdd(&segsum[s * HEADS + h], x);
}

// ---------------------------------------------------------------- segsum -> 1/segsum (in place)
__global__ void k_rcp(float* __restrict__ segsum) {
    int i = blockIdx.x * blockDim.x + threadIdx.x;
    if (i < N_NODES * HEADS) segsum[i] = 1.0f / segsum[i];
}

// ---------------------------------------------------------------- WMMA: x_j tile * coeff -> scatter
__global__ __launch_bounds__(256, 2)
void k_agg(const float* __restrict__ ea, const unsigned short* __restrict__ WTbf,
           const float* __restrict__ exw, const float* __restrict__ inv,
           const int* __restrict__ src, float* __restrict__ agg) {
    __shared__ unsigned short sA[128 * 128];   // 128 edges x 128 feat, bf16
    __shared__ unsigned short sB[128 * 128];   // W^T (N-major), bf16
    unsigned* sA32 = (unsigned*)sA;
    unsigned* sB32 = (unsigned*)sB;
    const int tid = threadIdx.x;

    // async global->LDS copy of B tile (32KB), 16B per lane per iteration
    unsigned ldsB0 = (unsigned)(uintptr_t)sB;
    #pragma unroll
    for (int it = 0; it < 8; ++it) {
        unsigned boff = (unsigned)(tid + it * 256) * 16u;
        async_b128_to_lds(ldsB0 + boff, boff, WTbf);
    }

    long e0 = (long)blockIdx.x * 128;
    for (int i = tid; i < 128 * 64; i += 256) {
        int row = i >> 6, cp = i & 63;
        long e = e0 + row;
        unsigned pack = 0u;
        if (e < E_EDGES) {
            float2 v = ((const float2*)(ea + e * DIM))[cp];
            pack = (unsigned)f2bf(v.x) | ((unsigned)f2bf(v.y) << 16);
        }
        sA32[i] = pack;
    }
    wait_async0();
    __syncthreads();

    const int wv = tid >> 5, lane = tid & 31;
    const int M = lane & 15, hi = lane >> 4;

    // A fragments (16x32 bf16 per K-step), ISA lane layout
    unsigned Au[4][8];
    #pragma unroll
    for (int ks = 0; ks < 4; ++ks)
        #pragma unroll
        for (int i = 0; i < 8; ++i) {
            int kk = 32 * ks + ((i < 4) ? 0 : 16) + hi * 8 + (i & 3) * 2;
            Au[ks][i] = sA32[(wv * 16 + M) * 64 + (kk >> 1)];
        }

    v8f accs[8];
    #pragma unroll
    for (int nt = 0; nt < 8; ++nt) accs[nt] = (v8f){};

    #pragma unroll
    for (int nt = 0; nt < 8; ++nt) {           // N-tile == head
        #pragma unroll
        for (int ks = 0; ks < 4; ++ks) {
            FragBF a, b;
            #pragma unroll
            for (int i = 0; i < 8; ++i) a.u[i] = Au[ks][i];
            #pragma unroll
            for (int i = 0; i < 8; ++i) {      // B 32x16 bf16, K-pair contiguous
                int kk = 32 * ks + ((i < 4) ? 0 : 16) + hi * 8 + (i & 3) * 2;
                b.u[i] = sB32[(nt * 16 + M) * 64 + (kk >> 1)];
            }
            accs[nt] = __builtin_amdgcn_wmma_f32_16x16x32_bf16(
                false, a.v, false, b.v, (short)0, accs[nt], false, false);
        }
    }

    // epilogue: one pass per output row; coeff = ex * (1/segsum) (no division)
    #pragma unroll
    for (int vv = 0; vv < 8; ++vv) {           // D row = vv + 8*hi, col = M
        long e = e0 + wv * 16 + vv + hi * 8;
        if (e < E_EDGES) {
            int s = src[e];
            float4 exa = ((const float4*)(exw + e * HEADS))[0];
            float4 exb = ((const float4*)(exw + e * HEADS))[1];
            float4 iva = ((const float4*)(inv + (size_t)s * HEADS))[0];
            float4 ivb = ((const float4*)(inv + (size_t)s * HEADS))[1];
            float co[8] = { exa.x * iva.x, exa.y * iva.y, exa.z * iva.z, exa.w * iva.w,
                            exb.x * ivb.x, exb.y * ivb.y, exb.z * ivb.z, exb.w * ivb.w };
            float* base = agg + (size_t)s * DIM + M;
            #pragma unroll
            for (int nt = 0; nt < 8; ++nt)
                atomicAdd(base + nt * CH, accs[nt][vv] * co[nt]);
        }
    }
}

// ---------------------------------------------------------------- elu + residual GEMM (WMMA)
__global__ __launch_bounds__(256, 2)
void k_final(const float* __restrict__ ea, const unsigned short* __restrict__ WTresbf,
             float* __restrict__ out) {
    __shared__ unsigned short sA[128 * 128];
    __shared__ unsigned short sB[128 * 128];
    unsigned* sA32 = (unsigned*)sA;
    unsigned* sB32 = (unsigned*)sB;
    const int tid = threadIdx.x;

    unsigned ldsB0 = (unsigned)(uintptr_t)sB;
    #pragma unroll
    for (int it = 0; it < 8; ++it) {
        unsigned boff = (unsigned)(tid + it * 256) * 16u;
        async_b128_to_lds(ldsB0 + boff, boff, WTresbf);
    }

    int n0 = blockIdx.x * 128;
    for (int i = tid; i < 128 * 64; i += 256) {
        int row = i >> 6, cp = i & 63;
        int n = n0 + row;
        unsigned pack = 0u;
        if (n < N_NODES) {
            float2 v = ((const float2*)(ea + (size_t)n * DIM))[cp];
            pack = (unsigned)f2bf(v.x) | ((unsigned)f2bf(v.y) << 16);
        }
        sA32[i] = pack;
    }
    wait_async0();
    __syncthreads();

    const int wv = tid >> 5, lane = tid & 31;
    const int M = lane & 15, hi = lane >> 4;

    unsigned Au[4][8];
    #pragma unroll
    for (int ks = 0; ks < 4; ++ks)
        #pragma unroll
        for (int i = 0; i < 8; ++i) {
            int kk = 32 * ks + ((i < 4) ? 0 : 16) + hi * 8 + (i & 3) * 2;
            Au[ks][i] = sA32[(wv * 16 + M) * 64 + (kk >> 1)];
        }

    #pragma unroll
    for (int nt = 0; nt < 8; ++nt) {
        v8f acc = {};
        #pragma unroll
        for (int ks = 0; ks < 4; ++ks) {
            FragBF a, b;
            #pragma unroll
            for (int i = 0; i < 8; ++i) a.u[i] = Au[ks][i];
            #pragma unroll
            for (int i = 0; i < 8; ++i) {
                int kk = 32 * ks + ((i < 4) ? 0 : 16) + hi * 8 + (i & 3) * 2;
                b.u[i] = sB32[(nt * 16 + M) * 64 + (kk >> 1)];
            }
            acc = __builtin_amdgcn_wmma_f32_16x16x32_bf16(
                false, a.v, false, b.v, (short)0, acc, false, false);
        }
        #pragma unroll
        for (int vv = 0; vv < 8; ++vv) {
            int n = n0 + wv * 16 + vv + hi * 8;
            if (n < N_NODES) {
                int col = nt * CH + M;
                float a = out[(size_t)n * DIM + col];            // aggregated value
                float el = (a > 0.f) ? a : (__expf(a) - 1.0f);   // elu
                out[(size_t)n * DIM + col] = el + acc[vv];       // + residual GEMM
            }
        }
    }
}

// ---------------------------------------------------------------- host
extern "C" void kernel_launch(void* const* d_in, const int* in_sizes, int n_in,
                              void* d_out, int out_size, void* d_ws, size_t ws_size,
                              hipStream_t stream) {
    const float* ea   = (const float*)d_in[0];   // edge_attr  (E,128)
    const float* W    = (const float*)d_in[1];   // (128,128)
    const float* attL = (const float*)d_in[2];   // (1,8,16)
    const float* attR = (const float*)d_in[3];
    const float* Wres = (const float*)d_in[4];   // (128,128)
    const int*   eidx = (const int*)d_in[5];     // (2,E)
    const int* tgt = eidx;
    const int* src = eidx + E_EDGES;
    float* out = (float*)d_out;

    char* ws = (char*)d_ws;
    auto al = [](size_t x) { return (x + 255) & ~(size_t)255; };
    size_t o = 0;
    float* wl = (float*)(ws + o);            o = al(o + DIM * HEADS * 4);
    float* wr = (float*)(ws + o);            o = al(o + DIM * HEADS * 4);
    unsigned short* WTbf    = (unsigned short*)(ws + o); o = al(o + DIM * DIM * 2);
    unsigned short* WTresbf = (unsigned short*)(ws + o); o = al(o + DIM * DIM * 2);
    float* r      = (float*)(ws + o);        o = al(o + (size_t)N_NODES * HEADS * 4);
    float* segmax = (float*)(ws + o);        o = al(o + (size_t)N_NODES * HEADS * 4);
    float* segsum = (float*)(ws + o);        o = al(o + (size_t)N_NODES * HEADS * 4);
    float* alpha  = (float*)(ws + o);        o = al(o + (size_t)E_EDGES * HEADS * 4);
    (void)ws_size; (void)in_sizes; (void)n_in; (void)out_size;

    k_init<<<(N_NODES * DIM + 255) / 256, 256, 0, stream>>>(out, segmax, segsum);
    k_prep<<<1, 128, 0, stream>>>(W, Wres, attL, attR, wl, wr, WTbf, WTresbf);
    k_node_r<<<(N_NODES * 4 + 255) / 256, 256, 0, stream>>>(ea, wr, r);
    k_alpha<<<(E_EDGES * 4 + 255) / 256, 256, 0, stream>>>(ea, wl, r, tgt, src, alpha, segmax);
    k_ex<<<(E_EDGES * HEADS + 255) / 256, 256, 0, stream>>>(src, alpha, segmax, segsum);
    k_rcp<<<(N_NODES * HEADS + 255) / 256, 256, 0, stream>>>(segsum);
    k_agg<<<(E_EDGES + 127) / 128, 256, 0, stream>>>(ea, WTbf, alpha, segsum, src, out);
    k_final<<<(N_NODES + 127) / 128, 256, 0, stream>>>(ea, WTresbf, out);
}
